// PatchEmbed_81982335746460
// MI455X (gfx1250) — matrix operand order repaired and checked by reading, not compiled, not run on hardware
//
#include <hip/hip_runtime.h>
#include <hip/hip_bf16.h>

// ---------------- problem constants ----------------
#define BB 16
#define OO 3
#define TT 8
#define CIN 3
#define IMG 224
#define PP 16
#define DD 768
#define HF 14
#define SS_ 7
#define MM (BB*TT*HF*HF)      // 25088 rows of the patch-embed GEMM
#define KK (CIN*PP*PP)        // 768
#define KSTEPS (KK/32)        // 24
#define HKS (KSTEPS/2)        // 12 (k-steps staged in LDS per phase)
#define NTILES (DD/16)        // 48
#define NBLOCKS (NTILES/4)    // 12 (each wave does a 16x64 strip)

typedef __attribute__((ext_vector_type(16))) __bf16 v16bf;
typedef __attribute__((ext_vector_type(8)))  float  v8f;

__device__ __forceinline__ unsigned short f2bf_bits(float f) {
    unsigned int u = __float_as_uint(f);
    unsigned int r = u + 0x7FFFu + ((u >> 16) & 1u);   // round-to-nearest-even
    return (unsigned short)(r >> 16);
}

// ---------------------------------------------------------------------------
// Pack proj_w (D x K, flat [d][k]) into WMMA B-fragment order:
// frag = nt*KSTEPS + ks ; per fragment 32 lanes x 16 bf16 contiguous per lane.
// B 32x16 layout: lanes 0-15 hold K=0..7,16..23, lanes 16-31 K=8..15,24..31 (elem e<8 -> K=k0+8h+e, e>=8 -> K=k0+8h+e+8).
// ---------------------------------------------------------------------------
__global__ void k_packw(const float* __restrict__ pw, unsigned short* __restrict__ bfrag) {
    int g = blockIdx.x * 256 + threadIdx.x;
    if (g >= DD * KK) return;
    int e    = g & 15;
    int lane = (g >> 4) & 31;
    int frag = g >> 9;
    int ks = frag % KSTEPS;
    int nt = frag / KSTEPS;
    int kk = ks * 32 + ((lane >= 16) ? 8 : 0) + ((e < 8) ? e : e + 8);
    int d  = nt * 16 + (lane & 15);
    bfrag[g] = f2bf_bits(pw[(size_t)d * KK + kk]);
}

// ---------------------------------------------------------------------------
// hidden[r][j] = relu( (boxes[r]/224) @ w1 ),  r = (b*T+t)*O + o,  384x384, K=4
// ---------------------------------------------------------------------------
__global__ void k_emb1(const float* __restrict__ meta, const float* __restrict__ w1,
                       float* __restrict__ hidden) {
    int idx = blockIdx.x * 256 + threadIdx.x;
    if (idx >= 384 * 384) return;
    int j = idx % 384;
    int r = idx / 384;
    int o = r % OO, t = (r / OO) % TT, b = r / (OO * TT);
    const float* mp = meta + ((size_t)(b * OO + o) * TT + t) * 4;
    float acc = 0.f;
#pragma unroll
    for (int i = 0; i < 4; ++i)
        acc += (mp[i] * (1.0f / 224.0f)) * w1[i * 384 + j];
    hidden[(size_t)r * 384 + j] = fmaxf(acc, 0.0f);
}

// ---------------------------------------------------------------------------
// emb[r][d] = box_categories[t][o][d] + relu( hidden[r] @ w2 ),  384x768, K=384
// ---------------------------------------------------------------------------
__global__ __launch_bounds__(256) void k_emb2(const float* __restrict__ hidden,
                                              const float* __restrict__ w2,
                                              const float* __restrict__ bc,
                                              float* __restrict__ emb) {
    __shared__ float hs[384];
    int r = blockIdx.x;
    for (int j = threadIdx.x; j < 384; j += 256) hs[j] = hidden[(size_t)r * 384 + j];
    __syncthreads();
    int o = r % OO, t = (r / OO) % TT;
    for (int d = threadIdx.x; d < DD; d += 256) {
        float acc = 0.f;
        for (int j = 0; j < 384; ++j)
            acc = fmaf(hs[j], w2[(size_t)j * DD + d], acc);
        emb[(size_t)r * DD + d] = bc[(size_t)(t * OO + o) * DD + d] + fmaxf(acc, 0.0f);
    }
}

// ---------------------------------------------------------------------------
// Patch-embed GEMM: C[m][d] = sum_k A[m][k]*W[d][k] + proj_b[d]
// m = ((b*T+t)*HF+h)*HF+w, k = c*256 + p*16 + q.
// Block = 8 waves sharing one 64-col N strip; B fragments staged in LDS in
// two 48 KB K-half phases; each wave computes a 16x64 strip (4 accumulators).
// A fragments loaded directly from x (two row-pairs of float4), pipelined one
// k-step ahead with prefetch two steps ahead.
// ---------------------------------------------------------------------------
__global__ __launch_bounds__(256) void k_gemm(const float* __restrict__ x,
                                              const float* __restrict__ proj_b,
                                              const unsigned short* __restrict__ bfrag_raw,
                                              float* __restrict__ C) {
    // 4 n-tiles * HKS k-steps * 512 shorts = 24576 shorts = 48 KB
    __shared__ __align__(32) unsigned short bsh[4 * HKS * 512];

    int warp = threadIdx.x >> 5;
    int lane = threadIdx.x & 31;
    int nb = blockIdx.x % NBLOCKS;    // this block's 64-wide N strip
    int mg = blockIdx.x / NBLOCKS;    // m-tile group (8 tiles, one per wave)
    int mt = mg * 8 + warp;
    int m0 = mt * 16;
    int mrow = m0 + (lane & 15);
    int w = mrow % HF;
    int tmp = mrow / HF;
    int h = tmp % HF; tmp /= HF;
    int t = tmp % TT;
    int b = tmp / TT;
    int hi = lane >> 4;               // 0/1: which K half this lane owns

    const int xplane = IMG * IMG;     // 50176

    // per-lane x address for a given k-step
    auto xaddr = [&](int ks) -> const float* {
        int base = ks * 32 + hi * 8;
        int c   = base >> 8;
        int rem = base & 255;
        int p   = rem >> 4;
        int q0  = rem & 15;           // 0 or 8
        return x + (size_t)((b * CIN + c) * TT + t) * xplane
                 + (h * PP + p) * IMG + (w * PP + q0);
    };

    v8f zero = {0.f,0.f,0.f,0.f,0.f,0.f,0.f,0.f};
    v8f acc0 = zero, acc1 = zero, acc2 = zero, acc3 = zero;

    const v16bf* bls = (const v16bf*)bsh;

    // --- A prologue: loads for ks = 0 ---
    const float* xa = xaddr(0);
    float4 f0 = *(const float4*)(xa);
    float4 f1 = *(const float4*)(xa + 4);
    float4 f2 = *(const float4*)(xa + IMG);
    float4 f3 = *(const float4*)(xa + IMG + 4);

    for (int half = 0; half < 2; ++half) {
        if (half) __syncthreads();    // all waves done reading previous phase
        // cooperative copy: 4 chunks of HKS*512 shorts (12 KB) each
        {
            const uint4* src = (const uint4*)(bfrag_raw
                               + ((size_t)(nb * 4) * KSTEPS + half * HKS) * 512);
            uint4* dst = (uint4*)bsh;
            const int chunk4 = HKS * 512 / 8;             // uint4 per chunk = 768
            const int tilestride4 = KSTEPS * 512 / 8;     // uint4 stride between n-tiles
#pragma unroll
            for (int s = 0; s < 4; ++s) {
                for (int i = threadIdx.x; i < chunk4; i += 256)
                    dst[s * chunk4 + i] = src[(size_t)s * tilestride4 + i];
            }
        }
        __syncthreads();

        for (int ks2 = 0; ks2 < HKS; ++ks2) {
            int ks = half * HKS + ks2;
            // convert current A floats to a bf16 fragment (v_cvt_pk_bf16_f32)
            union { v16bf v; __bf16 hh[16]; } a;
            a.hh[0]  = (__bf16)f0.x; a.hh[1]  = (__bf16)f0.y; a.hh[2]  = (__bf16)f0.z; a.hh[3]  = (__bf16)f0.w;
            a.hh[4]  = (__bf16)f1.x; a.hh[5]  = (__bf16)f1.y; a.hh[6]  = (__bf16)f1.z; a.hh[7]  = (__bf16)f1.w;
            a.hh[8]  = (__bf16)f2.x; a.hh[9]  = (__bf16)f2.y; a.hh[10] = (__bf16)f2.z; a.hh[11] = (__bf16)f2.w;
            a.hh[12] = (__bf16)f3.x; a.hh[13] = (__bf16)f3.y; a.hh[14] = (__bf16)f3.z; a.hh[15] = (__bf16)f3.w;

            // issue next k-step's A loads before the WMMAs (dummy reload at end)
            int ksn = (ks < KSTEPS - 1) ? ks + 1 : ks;
            const float* xn = xaddr(ksn);
            __builtin_prefetch(xaddr((ks + 2 < KSTEPS) ? ks + 2 : ks), 0, 3);
            f0 = *(const float4*)(xn);
            f1 = *(const float4*)(xn + 4);
            f2 = *(const float4*)(xn + IMG);
            f3 = *(const float4*)(xn + IMG + 4);

            // B fragments from LDS (ds_load_b128 pairs)
            v16bf b0 = bls[(0 * HKS + ks2) * 32 + lane];
            v16bf b1 = bls[(1 * HKS + ks2) * 32 + lane];
            v16bf b2 = bls[(2 * HKS + ks2) * 32 + lane];
            v16bf b3 = bls[(3 * HKS + ks2) * 32 + lane];

            acc0 = __builtin_amdgcn_wmma_f32_16x16x32_bf16(false, a.v, false, b0, (short)0, acc0, false, false);
            acc1 = __builtin_amdgcn_wmma_f32_16x16x32_bf16(false, a.v, false, b1, (short)0, acc1, false, false);
            acc2 = __builtin_amdgcn_wmma_f32_16x16x32_bf16(false, a.v, false, b2, (short)0, acc2, false, false);
            acc3 = __builtin_amdgcn_wmma_f32_16x16x32_bf16(false, a.v, false, b3, (short)0, acc3, false, false);
        }
    }

    // C/D layout: VGPR r -> row m0 + r + 8*hi, column = n0 + (lane&15)
    int mbase = m0 + hi * 8;
    v8f accs[4] = {acc0, acc1, acc2, acc3};
#pragma unroll
    for (int s = 0; s < 4; ++s) {
        int d = nb * 64 + s * 16 + (lane & 15);
        float bias = proj_b[d];
#pragma unroll
        for (int r = 0; r < 8; ++r) {
            C[(size_t)(mbase + r) * DD + d] = accs[s][r] + bias;
        }
    }
}

// ---------------------------------------------------------------------------
// ROI bilinear crop from feat (= C[m][d]) + emb add + final layout:
// out[((b*O+o)*T + t)*49 + ss][d]
// ---------------------------------------------------------------------------
__global__ __launch_bounds__(256) void k_roi(const float* __restrict__ Cfeat,
                                             const float* __restrict__ meta,
                                             const float* __restrict__ emb,
                                             float* __restrict__ out) {
    int pos = blockIdx.x;             // ((bo)*T + t)*49 + ss
    int ss = pos % 49;
    int bt_ = pos / 49;
    int t = bt_ % TT;
    int bo = bt_ / TT;
    int o = bo % OO, b = bo / OO;
    int sy = ss / SS_, sx = ss % SS_;

    const float* mp = meta + ((size_t)(b * OO + o) * TT + t) * 4;
    const float scale = (float)HF / 224.0f;
    float x1 = mp[0] * scale, y1 = mp[1] * scale;
    float x2 = mp[2] * scale, y2 = mp[3] * scale;
    float cy = (sy + 0.5f) / (float)SS_;
    float cx = (sx + 0.5f) / (float)SS_;
    float yg = y1 + cy * (y2 - y1) - 0.5f;
    float xg = x1 + cx * (x2 - x1) - 0.5f;
    float y0f = floorf(yg), x0f = floorf(xg);
    float wy = yg - y0f, wx = xg - x0f;
    int y0  = min(max((int)y0f, 0), HF - 1);
    int y1i = min(y0 + 1, HF - 1);
    int x0  = min(max((int)x0f, 0), HF - 1);
    int x1i = min(x0 + 1, HF - 1);

    int bt = b * TT + t;
    size_t base = (size_t)bt * (HF * HF);
    const float* p00 = Cfeat + (base + y0  * HF + x0 ) * DD;
    const float* p01 = Cfeat + (base + y0  * HF + x1i) * DD;
    const float* p10 = Cfeat + (base + y1i * HF + x0 ) * DD;
    const float* p11 = Cfeat + (base + y1i * HF + x1i) * DD;
    float w00 = (1.f - wy) * (1.f - wx);
    float w01 = (1.f - wy) * wx;
    float w10 = wy * (1.f - wx);
    float w11 = wy * wx;

    const float* ep = emb + (size_t)((b * TT + t) * OO + o) * DD;
    float* op = out + (size_t)pos * DD;
    for (int d = threadIdx.x; d < DD; d += 256) {
        op[d] = p00[d]*w00 + p01[d]*w01 + p10[d]*w10 + p11[d]*w11 + ep[d];
    }
}

// ---------------------------------------------------------------------------
extern "C" void kernel_launch(void* const* d_in, const int* in_sizes, int n_in,
                              void* d_out, int out_size, void* d_ws, size_t ws_size,
                              hipStream_t stream) {
    const float* x      = (const float*)d_in[0];   // (16,3,8,224,224)
    const float* meta   = (const float*)d_in[1];   // (16,3,8,4)
    const float* proj_w = (const float*)d_in[2];   // (768,3,16,16) flat = [d][k]
    const float* proj_b = (const float*)d_in[3];   // (768,)
    const float* bc     = (const float*)d_in[4];   // (8,3,768)
    const float* w1     = (const float*)d_in[5];   // (4,384)
    const float* w2     = (const float*)d_in[6];   // (384,768)
    float* out = (float*)d_out;                    // (48,8,49,768)

    // workspace layout
    char* ws = (char*)d_ws;
    float*          Cfeat  = (float*)ws;                               // 25088*768*4 = 77,070,336 B
    unsigned short* bfrag  = (unsigned short*)(ws + 77070336);         // 768*768*2   =  1,179,648 B
    float*          hidden = (float*)(ws + 77070336 + 1179648);        // 384*384*4   =    589,824 B
    float*          emb    = (float*)(ws + 77070336 + 1179648 + 589824); // 384*768*4 =  1,179,648 B

    // 1) pack proj_w into WMMA B-fragment layout (bf16)
    k_packw<<<(DD * KK + 255) / 256, 256, 0, stream>>>(proj_w, bfrag);
    // 2) box MLP
    k_emb1<<<(384 * 384 + 255) / 256, 256, 0, stream>>>(meta, w1, hidden);
    k_emb2<<<384, 256, 0, stream>>>(hidden, w2, bc, emb);
    // 3) patch-embed GEMM via WMMA: 196 m-groups x 12 n-strips = 2352 blocks
    k_gemm<<<(MM / 16 / 8) * NBLOCKS, 256, 0, stream>>>(x, proj_b, bfrag, Cfeat);
    // 4) ROI crop + emb add + final layout: one block per output position
    k_roi<<<(BB * OO) * TT * 49, 256, 0, stream>>>(Cfeat, meta, emb, out);
}
